// CrossLayerAttention_309237645906
// MI455X (gfx1250) — matrix operand order repaired and checked by reading, not compiled, not run on hardware
//
#include <hip/hip_runtime.h>

// ---------------------------------------------------------------------------
// CDNA5 (gfx1250) attention block:
//   0) transpose-convert K,V f32 -> f16 (once; overlaid on d_out scratch)
//   1) qproj GEMM (f16 WMMA, f32 accum)
//   2) causal flash attention, async-to-LDS double-buffered K/V staging
//   3) out-proj GEMM
// ---------------------------------------------------------------------------

typedef __attribute__((ext_vector_type(16))) _Float16 v16h;
typedef __attribute__((ext_vector_type(8)))  float    v8f;
typedef __attribute__((ext_vector_type(4)))  int      v4i;

struct U4 { unsigned int a, b, c, d; };
struct U2 { unsigned int a, b; };
struct F4 { float x, y, z, w; };

union FragH { v16h h; U4 q[2]; };

constexpr int B_  = 2;
constexpr int SQ  = 2048;
constexpr int SK  = 2048;
constexpr int H   = 2048;
constexpr int NH  = 16;
constexpr int HD  = 128;
constexpr float SCALE = 0.0883883476483184f; // 1/sqrt(128)

__device__ inline v8f zero8() {
  v8f z = {0.f, 0.f, 0.f, 0.f, 0.f, 0.f, 0.f, 0.f};
  return z;
}

// A-fragment (16x32 f16): lanes 0-15 = M rows, hi = lane>>4 selects K half.
__device__ inline v16h ld_frag_a(const _Float16* base, int pitch, int row, int hi, int k0) {
  const _Float16* r = base + (size_t)row * pitch + k0;
  FragH f;
  f.q[0] = *(const U4*)(r + hi * 8);
  f.q[1] = *(const U4*)(r + 16 + hi * 8);
  return f.h;
}

// B-fragment (32x16 f16): lanes 0-15 = N cols, hi = lane>>4 selects K half (same order as A).
__device__ inline v16h ld_frag_b(const _Float16* base, int pitch, int col, int hi, int k0) {
  const _Float16* r = base + (size_t)col * pitch + k0;
  FragH f;
  f.q[0] = *(const U4*)(r + hi * 8);
  f.q[1] = *(const U4*)(r + 16 + hi * 8);
  return f.h;
}

__device__ inline v8f wmma_f16(v16h a, v16h b, v8f c) {
  return __builtin_amdgcn_wmma_f32_16x16x32_f16(false, a, false, b, (short)0, c,
                                                false, false);
}

__device__ inline float redmax16(float v) {
#pragma unroll
  for (int m = 1; m <= 8; m <<= 1) v = fmaxf(v, __shfl_xor(v, m, 32));
  return v;
}
__device__ inline float redsum16(float v) {
#pragma unroll
  for (int m = 1; m <= 8; m <<= 1) v += __shfl_xor(v, m, 32);
  return v;
}

// --- CDNA5 async copy: 16 bytes global -> LDS, tracked by ASYNCcnt ---------
__device__ inline void async_b128(const _Float16* g, _Float16* l) {
#if __has_builtin(__builtin_amdgcn_global_load_async_to_lds_b128)
  typedef __attribute__((address_space(1))) v4i* gv4i_p;
  typedef __attribute__((address_space(3))) v4i* lv4i_p;
  __builtin_amdgcn_global_load_async_to_lds_b128(
      (gv4i_p)(void*)g, (lv4i_p)(void*)l, 0, 0);
#else
  unsigned loff =
      (unsigned)(__UINTPTR_TYPE__)(__attribute__((address_space(3))) void*)l;
  asm volatile("global_load_async_to_lds_b128 %0, %1, off"
               :: "v"(loff), "v"(g) : "memory");
#endif
}

template <int N>
__device__ inline void wait_asynccnt() {
#if __has_builtin(__builtin_amdgcn_s_wait_asynccnt)
  __builtin_amdgcn_s_wait_asynccnt((unsigned short)N);
#else
  asm volatile("s_wait_asynccnt %0" :: "i"(N) : "memory");
#endif
}

// ---------------------------------------------------------------------------
// Transpose-convert: in f32 [z][R][C] -> out f16 [z][C][R]
// ---------------------------------------------------------------------------
__global__ __launch_bounds__(256) void tconv_kernel(const float* __restrict__ in,
                                                    _Float16* __restrict__ out,
                                                    int R, int C) {
  __shared__ float tile[32][33];
  const int z = blockIdx.z;
  const int r0 = blockIdx.y * 32, c0 = blockIdx.x * 32;
  const float* src = in + (size_t)z * R * C;
  _Float16* dst = out + (size_t)z * R * C;
  const int tx = threadIdx.x & 31, ty = threadIdx.x >> 5; // 32 x 8
#pragma unroll
  for (int i = 0; i < 32; i += 8)
    tile[ty + i][tx] = src[(size_t)(r0 + ty + i) * C + c0 + tx];
  __syncthreads();
#pragma unroll
  for (int i = 0; i < 32; i += 8)
    dst[(size_t)(c0 + ty + i) * R + r0 + tx] = (_Float16)tile[tx][ty + i];
}

// ---------------------------------------------------------------------------
// GEMM:  C[M=4096, N=2048] = A[M, K=2048] * W[N, K]^T + bias
// OUT_MODE 0: f32 linear output.  OUT_MODE 1: f16 scatter to [B*NH, SQ, HD].
// ---------------------------------------------------------------------------
template <int OUT_MODE, bool A_F16>
__global__ __launch_bounds__(256) void gemm_kernel(const void* __restrict__ Aptr,
                                                   const float* __restrict__ W,
                                                   const float* __restrict__ bias,
                                                   _Float16* __restrict__ out_h,
                                                   float* __restrict__ out_f) {
  constexpr int K = H;
  constexpr int N = H;
  constexpr int BK = 64;
  constexpr int PITCH = BK + 8;
  __shared__ _Float16 As[128 * PITCH];
  __shared__ _Float16 Bs[128 * PITCH];

  const int t = threadIdx.x;
  const int lane = t & 31;
  const int wave = t >> 5;
  const int lo = lane & 15, hi = lane >> 4;
  const int wm = wave >> 1, wn = wave & 1;
  const int m0 = blockIdx.y * 128;
  const int n0 = blockIdx.x * 128;

  v8f acc[2][4];
#pragma unroll
  for (int i = 0; i < 2; ++i)
#pragma unroll
    for (int j = 0; j < 4; ++j) acc[i][j] = zero8();

  const float*    Af = (const float*)Aptr;
  const _Float16* Ah = (const _Float16*)Aptr;

  for (int bk = 0; bk < K; bk += BK) {
#pragma unroll 2
    for (int p = 0; p < 8; ++p) {
      int flat = p * 1024 + t * 4;
      int row = flat >> 6;
      int k = flat & 63;
      if constexpr (A_F16) {
        *(U2*)&As[row * PITCH + k] =
            *(const U2*)(Ah + (size_t)(m0 + row) * K + bk + k);
      } else {
        F4 s = *(const F4*)(Af + (size_t)(m0 + row) * K + bk + k);
        As[row * PITCH + k + 0] = (_Float16)s.x;
        As[row * PITCH + k + 1] = (_Float16)s.y;
        As[row * PITCH + k + 2] = (_Float16)s.z;
        As[row * PITCH + k + 3] = (_Float16)s.w;
      }
      F4 wv = *(const F4*)(W + (size_t)(n0 + row) * K + bk + k);
      Bs[row * PITCH + k + 0] = (_Float16)wv.x;
      Bs[row * PITCH + k + 1] = (_Float16)wv.y;
      Bs[row * PITCH + k + 2] = (_Float16)wv.z;
      Bs[row * PITCH + k + 3] = (_Float16)wv.w;
    }
    __syncthreads();

#pragma unroll
    for (int ks = 0; ks < 2; ++ks) {
      const int k0 = ks * 32;
      v16h af[2], bf[4];
#pragma unroll
      for (int i = 0; i < 2; ++i)
        af[i] = ld_frag_a(As, PITCH, wm * 32 + i * 16 + lo, hi, k0);
#pragma unroll
      for (int j = 0; j < 4; ++j)
        bf[j] = ld_frag_b(Bs, PITCH, wn * 64 + j * 16 + lo, hi, k0);
#pragma unroll
      for (int i = 0; i < 2; ++i)
#pragma unroll
        for (int j = 0; j < 4; ++j)
          acc[i][j] = wmma_f16(af[i], bf[j], acc[i][j]);
    }
    __syncthreads();
  }

#pragma unroll
  for (int i = 0; i < 2; ++i) {
#pragma unroll
    for (int j = 0; j < 4; ++j) {
      int col = n0 + wn * 64 + j * 16 + lo;
      float bv = bias[col];
#pragma unroll
      for (int r = 0; r < 8; ++r) {
        int row = m0 + wm * 32 + i * 16 + r + hi * 8;
        float v = acc[i][j][r] + bv;
        if constexpr (OUT_MODE == 0) {
          out_f[(size_t)row * N + col] = v;
        } else {
          int b = row >> 11, sq = row & (SQ - 1);
          int nh = col >> 7, hd = col & (HD - 1);
          out_h[(((size_t)(b * NH + nh)) * SQ + sq) * HD + hd] = (_Float16)v;
        }
      }
    }
  }
}

// ---------------------------------------------------------------------------
// Causal flash attention.  Grid (SQ/64, B*NH), block 128 thr = 4 waves.
// K/V pre-converted to f16 and pre-transposed:
//   k16: [B*NH, SK, HD] (hd contiguous)  -> Ks[sk][hd]  (direct async rows)
//   v16: [B*NH, HD, SK] (sk contiguous)  -> Vs[hd][sk]  (direct async rows)
// Double-buffered LDS staging via ASYNCcnt.
// ---------------------------------------------------------------------------
__global__ __launch_bounds__(128) void attn_kernel(const _Float16* __restrict__ qw,
                                                   const _Float16* __restrict__ k16,
                                                   const _Float16* __restrict__ v16,
                                                   _Float16* __restrict__ attn_out) {
  constexpr int KP = HD + 8;   // 136
  constexpr int VP = 64 + 8;   // 72
  constexpr int PP = 64 + 8;   // 72
  __shared__ _Float16 Ks[2][64 * KP];
  __shared__ _Float16 Vs[2][HD * VP];
  __shared__ _Float16 Ps[4][16 * PP];

  const int t = threadIdx.x;
  const int lane = t & 31, w = t >> 5;
  const int lo = lane & 15, hi = lane >> 4;
  const int bh = blockIdx.y;
  const int b = bh >> 4, nh = bh & 15;
  const int q0 = blockIdx.x * 64;
  const int qrow = q0 + w * 16 + lo;

  // issue one K/V block stage: 16 async b128 per thread (8 K + 8 V)
  auto issue_stage = [&](int buf, int kb) {
    const int sk0 = kb * 64;
    const _Float16* kgl = k16 + ((size_t)bh * SK + sk0) * HD;
#pragma unroll
    for (int c = 0; c < 8; ++c) {
      int g = c * 128 + t;
      int sk = g >> 4, hd = (g & 15) * 8;
      async_b128(kgl + (size_t)sk * HD + hd, &Ks[buf][sk * KP + hd]);
    }
    const _Float16* vgl = v16 + (size_t)bh * HD * SK + sk0;
#pragma unroll
    for (int c = 0; c < 8; ++c) {
      int g = c * 128 + t;
      int hd = g >> 3, sk = (g & 7) * 8;
      async_b128(vgl + (size_t)hd * SK + sk, &Vs[buf][hd * VP + sk]);
    }
  };

  // Q fragments for this wave's 16 rows (resident): 4 K-steps over HD=128
  v16h aq[4];
  const _Float16* qbase = qw + ((size_t)bh * SQ + qrow) * HD;
#pragma unroll
  for (int s = 0; s < 4; ++s) aq[s] = ld_frag_a(qbase, 0, 0, hi, s * 32);

  v8f o[8];
  float m_run[8], l_run[8];
#pragma unroll
  for (int h = 0; h < 8; ++h) o[h] = zero8();
#pragma unroll
  for (int r = 0; r < 8; ++r) { m_run[r] = -1e30f; l_run[r] = 0.f; }

  const int nkb = (q0 >> 6) + 1;   // causal: only key blocks up to the diagonal
  issue_stage(0, 0);

  for (int kb = 0; kb < nkb; ++kb) {
    const int cur = kb & 1;
    // prefetch next block while current computes; wait for current's 16 ops
    if (kb + 1 < nkb) {
      issue_stage(cur ^ 1, kb + 1);
      wait_asynccnt<16>();
    } else {
      wait_asynccnt<0>();
    }
    __syncthreads();

    const int sk0 = kb * 64;
    const _Float16* ksb = &Ks[cur][0];
    const _Float16* vsb = &Vs[cur][0];

    // S = Q @ K^T  (16 x 64, f32 accum)
    v8f sacc[4];
#pragma unroll
    for (int j = 0; j < 4; ++j) sacc[j] = zero8();
#pragma unroll
    for (int j = 0; j < 4; ++j)
#pragma unroll
      for (int s = 0; s < 4; ++s) {
        v16h bk = ld_frag_b(ksb, KP, j * 16 + lo, hi, s * 32);
        sacc[j] = wmma_f16(aq[s], bk, sacc[j]);
      }

    // online softmax (rows span a 16-lane half-group in C layout)
    _Float16* pw = &Ps[w][0];
#pragma unroll
    for (int r = 0; r < 8; ++r) {
      const int row = q0 + w * 16 + r + hi * 8;
      float v[4];
      float rmax = -1e30f;
#pragma unroll
      for (int j = 0; j < 4; ++j) {
        int col = sk0 + j * 16 + lo;
        float x = sacc[j][r] * SCALE;
        x = (col <= row) ? x : -1e30f;   // causal mask, analytic
        v[j] = x;
        rmax = fmaxf(rmax, x);
      }
      rmax = redmax16(rmax);
      float m_new = fmaxf(m_run[r], rmax);
      float corr = __expf(m_run[r] - m_new);
      float rsum = 0.f;
#pragma unroll
      for (int j = 0; j < 4; ++j) {
        float pj = __expf(v[j] - m_new);
        rsum += pj;
        pw[(r + hi * 8) * PP + j * 16 + lo] = (_Float16)pj;
      }
      rsum = redsum16(rsum);
      l_run[r] = l_run[r] * corr + rsum;
      m_run[r] = m_new;
#pragma unroll
      for (int h = 0; h < 8; ++h) o[h][r] *= corr;
    }

    // O += P @ V   (K = 64 keys -> 2 WMMA K-steps; 8 HD subtiles)
#pragma unroll
    for (int s2 = 0; s2 < 2; ++s2) {
      v16h pa = ld_frag_a(pw, PP, lo, hi, s2 * 32);
#pragma unroll
      for (int h = 0; h < 8; ++h) {
        v16h bv = ld_frag_b(vsb, VP, h * 16 + lo, hi, s2 * 32);
        o[h] = wmma_f16(pa, bv, o[h]);
      }
    }
    __syncthreads();
  }

  // normalize and write [B*SQ, H] f16
#pragma unroll
  for (int h = 0; h < 8; ++h) {
    int colh = nh * HD + h * 16 + lo;
#pragma unroll
    for (int r = 0; r < 8; ++r) {
      int row = q0 + w * 16 + r + hi * 8;
      float vout = o[h][r] / l_run[r];
      attn_out[((size_t)b * SQ + row) * H + colh] = (_Float16)vout;
    }
  }
}

// ---------------------------------------------------------------------------
extern "C" void kernel_launch(void* const* d_in, const int* in_sizes, int n_in,
                              void* d_out, int out_size, void* d_ws, size_t ws_size,
                              hipStream_t stream) {
  (void)in_sizes; (void)n_in; (void)out_size; (void)ws_size;
  const float* hidden = (const float*)d_in[0];
  const float* key    = (const float*)d_in[1];
  const float* value  = (const float*)d_in[2];
  // d_in[3] additive causal mask: recomputed analytically, not read
  const float* w_q    = (const float*)d_in[4];
  const float* b_q    = (const float*)d_in[5];
  const float* w_proj = (const float*)d_in[6];
  const float* b_proj = (const float*)d_in[7];
  float* out = (float*)d_out;

  _Float16* q_ws    = (_Float16*)d_ws;                       // 16.8 MB
  _Float16* attn_ws = q_ws + (size_t)B_ * NH * SQ * HD;      // 16.8 MB

  // f16 K/V scratch overlaid on d_out (33.5 MB); fully rewritten by kernel 3.
  _Float16* k16 = (_Float16*)d_out;
  _Float16* v16 = k16 + (size_t)B_ * NH * SK * HD;

  // 0) K [bh][HD][SK] f32 -> k16 [bh][SK][HD] f16 ; V [bh][SK][HD] -> v16 [bh][HD][SK]
  tconv_kernel<<<dim3(SK / 32, HD / 32, B_ * NH), 256, 0, stream>>>(key, k16, HD, SK);
  tconv_kernel<<<dim3(HD / 32, SK / 32, B_ * NH), 256, 0, stream>>>(value, v16, SK, HD);

  dim3 gblk(H / 128, (B_ * SQ) / 128);   // (16, 32)

  // 1) q = hidden @ w_q^T + b_q  -> f16 scatter to [B*NH, SQ, HD]
  gemm_kernel<1, false><<<gblk, 256, 0, stream>>>(hidden, w_q, b_q, q_ws, nullptr);

  // 2) causal flash attention -> [B*SQ, H] f16
  attn_kernel<<<dim3(SQ / 64, B_ * NH), 128, 0, stream>>>(q_ws, k16, v16, attn_ws);

  // 3) out = attn @ w_proj^T + b_proj -> f32 d_out
  gemm_kernel<0, true><<<gblk, 256, 0, stream>>>(attn_ws, w_proj, b_proj, nullptr, out);
}
